// ImprovedHeteroGraphSAGE_35416300323550
// MI455X (gfx1250) — matrix operand order repaired and checked by reading, not compiled, not run on hardware
//
#include <hip/hip_runtime.h>
#include <hip/hip_bf16.h>

typedef __attribute__((ext_vector_type(16))) _Float16 v16h;
typedef __attribute__((ext_vector_type(2)))  _Float16 v2h;
typedef __attribute__((ext_vector_type(8))) float v8f;
typedef __attribute__((ext_vector_type(2))) float v2f;

#define HID 128
#define EPSV 1e-5f
#define NTEAM 30

// Build a 32x16 B fragment from an LDS weight row chunk (row-major [n][32]).
// Contiguous k order used here: lane gives n = lane%16; reg r holds k = (lane>=16?16:0)+2r, 2r+1.
__device__ __forceinline__ void load_bfrag(v16h& bf, const _Float16* __restrict__ w) {
#pragma unroll
  for (int r = 0; r < 8; ++r) {
    v2h p = *(const v2h*)(w + 2 * r);
    bf[2 * r] = p[0];
    bf[2 * r + 1] = p[1];
  }
}

// ---------------------------------------------------------------- utilities
__global__ void f32_to_f16_kernel(const float* __restrict__ in,
                                  _Float16* __restrict__ out, int n) {
  int i = blockIdx.x * 256 + threadIdx.x;
  if (i < n) out[i] = (_Float16)in[i];
}

__global__ void count_kernel(const int* __restrict__ dst,
                             float* __restrict__ deg, int E) {
  int e = blockIdx.x * 256 + threadIdx.x;
  if (e < E) atomicAdd(deg + dst[e], 1.0f);
}

// ------------------------------------------------------------------- stems
__global__ void stem_kernel(const float* __restrict__ x, int din,
                            const float* __restrict__ W, const float* __restrict__ b,
                            const float* __restrict__ g, const float* __restrict__ be,
                            const float* __restrict__ mn, const float* __restrict__ vr,
                            _Float16* __restrict__ outH, int N) {
  int idx = blockIdx.x * 256 + threadIdx.x;
  int o = idx & (HID - 1);
  int n = idx >> 7;
  if (n >= N) return;
  const float* xr = x + (size_t)n * din;
  const float* wr = W + (size_t)o * din;
  float v = b[o];
  for (int i = 0; i < din; ++i) v += xr[i] * wr[i];
  float sc = g[o] * rsqrtf(vr[o] + EPSV);
  v = v * sc + (be[o] - mn[o] * sc);
  outH[(size_t)idx] = (_Float16)fmaxf(v, 0.0f);
}

// -------------------------------------------------- edge scatter (wave/edge)
__global__ void scatter_kernel(const _Float16* __restrict__ xsrc,
                               const int* __restrict__ src, const int* __restrict__ dst,
                               float* __restrict__ agg, int E) {
  int gid = blockIdx.x * 256 + threadIdx.x;
  int e = gid >> 5, lane = gid & 31;
  if (e >= E) return;
  int s = src[e], d = dst[e];
  const v2h* sp = (const v2h*)(xsrc + (size_t)s * HID) + lane * 2;
  v2h p0 = sp[0], p1 = sp[1];
  float* dp = agg + (size_t)d * HID + lane * 4;
  atomicAdd(dp + 0, (float)p0[0]);
  atomicAdd(dp + 1, (float)p0[1]);
  atomicAdd(dp + 2, (float)p1[0]);
  atomicAdd(dp + 3, (float)p1[1]);
}

// ------------------------------- team scatter: LDS pre-reduction then flush
__global__ __launch_bounds__(256) void scatter_team_kernel(
    const _Float16* __restrict__ xsrc,
    const int* __restrict__ src, const int* __restrict__ dst,
    float* __restrict__ agg, float* __restrict__ deg, int E) {
  __shared__ float ls[NTEAM * HID];
  __shared__ float ld[NTEAM];
  for (int i = threadIdx.x; i < NTEAM * HID; i += 256) ls[i] = 0.0f;
  if (threadIdx.x < NTEAM) ld[threadIdx.x] = 0.0f;
  __syncthreads();
  int lane = threadIdx.x & 31;
  int wave = (blockIdx.x * 256 + threadIdx.x) >> 5;
  int nw = gridDim.x * 8;
  for (int e = wave; e < E; e += nw) {
    int s = src[e], d = dst[e];
    const v2h* sp = (const v2h*)(xsrc + (size_t)s * HID) + lane * 2;
    v2h p0 = sp[0], p1 = sp[1];
    float* dp = ls + d * HID + lane * 4;
    atomicAdd(dp + 0, (float)p0[0]);
    atomicAdd(dp + 1, (float)p0[1]);
    atomicAdd(dp + 2, (float)p1[0]);
    atomicAdd(dp + 3, (float)p1[1]);
    if (lane == 0) atomicAdd(ld + d, 1.0f);
  }
  __syncthreads();
  for (int i = threadIdx.x; i < NTEAM * HID; i += 256)
    if (ls[i] != 0.0f) atomicAdd(agg + i, ls[i]);
  if (threadIdx.x < NTEAM && ld[threadIdx.x] != 0.0f)
    atomicAdd(deg + threadIdx.x, ld[threadIdx.x]);
}

// --------------------------------------- fused SAGE layer: WMMA dual GEMM
// out = relu(bn( (agg/deg)@Wl^T + bl + x@Wr^T ))
// one wave = 16 rows x 128 cols; K=128 in 4 LDS-staged chunks of 32.
// All 8 B fragments of a weight matrix are made live at once so the 16
// ds_load_b128 issue as one batch and dscnt waits stagger across the WMMAs.
__global__ __launch_bounds__(256) void sage_wmma_kernel(
    const float* __restrict__ aggsum, const float* __restrict__ deg,
    const _Float16* __restrict__ xdst,
    const _Float16* __restrict__ WlH, const _Float16* __restrict__ WrH,
    const float* __restrict__ bl,
    const float* __restrict__ g, const float* __restrict__ be,
    const float* __restrict__ mn, const float* __restrict__ vr,
    _Float16* __restrict__ outH, int N) {
  __shared__ _Float16 sWl[HID * 32];
  __shared__ _Float16 sWr[HID * 32];
  const int lane = threadIdx.x & 31;
  const int wave = threadIdx.x >> 5;
  const int rowbase = blockIdx.x * 128 + wave * 16;
  const int mrow = rowbase + (lane & 15);
  const int mload = (mrow < N) ? mrow : (N - 1);
  const float inv = 1.0f / fmaxf(deg[mload], 1.0f);
  const float* arow = aggsum + (size_t)mload * HID;
  const _Float16* xrow = xdst + (size_t)mload * HID;
  const int khi = (lane >= 16) ? 8 : 0;
  const int bhi = (lane >= 16) ? 16 : 0;
  const int nlo = lane & 15;

  v8f acc[8];
#pragma unroll
  for (int t = 0; t < 8; ++t) {
    v8f z = {0.f, 0.f, 0.f, 0.f, 0.f, 0.f, 0.f, 0.f};
    acc[t] = z;
  }

  for (int s = 0; s < 4; ++s) {
    __syncthreads();
    // stage W[:, s*32 .. s*32+31] for both matrices: 512 int4 each
    for (int i = threadIdx.x; i < HID * 4; i += 256) {
      int n = i >> 2;
      int kk = (i & 3) * 8;
      ((int4*)sWl)[i] = *(const int4*)(WlH + (size_t)n * HID + s * 32 + kk);
      ((int4*)sWr)[i] = *(const int4*)(WrH + (size_t)n * HID + s * 32 + kk);
    }
    __syncthreads();

    // A fragments (contiguous k = 16*(lane/16) + i, same order as B)
    v16h a1, a2;
#pragma unroll
    for (int r = 0; r < 8; ++r) {
      int k = s * 32 + 2 * khi + r * 2;
      v2f f = *(const v2f*)(arow + k);
      a1[2 * r] = (_Float16)(f[0] * inv);
      a1[2 * r + 1] = (_Float16)(f[1] * inv);
      v2h hx = *(const v2h*)(xrow + k);
      a2[2 * r] = hx[0];
      a2[2 * r + 1] = hx[1];
    }

    // pass 1: Wl with mean-aggregated A — batch all fragment loads first
    {
      v16h bf[8];
#pragma unroll
      for (int t = 0; t < 8; ++t)
        load_bfrag(bf[t], sWl + (t * 16 + nlo) * 32 + bhi);
#pragma unroll
      for (int t = 0; t < 8; ++t)
        acc[t] = __builtin_amdgcn_wmma_f32_16x16x32_f16(false, a1, false, bf[t],
                                                        (short)0, acc[t], false, false);
    }
    // pass 2: Wr with root A
    {
      v16h bf[8];
#pragma unroll
      for (int t = 0; t < 8; ++t)
        load_bfrag(bf[t], sWr + (t * 16 + nlo) * 32 + bhi);
#pragma unroll
      for (int t = 0; t < 8; ++t)
        acc[t] = __builtin_amdgcn_wmma_f32_16x16x32_f16(false, a2, false, bf[t],
                                                        (short)0, acc[t], false, false);
    }
  }

  // epilogue: + bias, BN, ReLU, f16 store (C layout: n=lane%16, m=base+(lane>=16?8:0)+r)
  const int mbase = rowbase + ((lane >= 16) ? 8 : 0);
#pragma unroll
  for (int t = 0; t < 8; ++t) {
    int n = t * 16 + nlo;
    float bb = bl[n];
    float sc = g[n] * rsqrtf(vr[n] + EPSV);
    float sh = be[n] - mn[n] * sc;
#pragma unroll
    for (int r = 0; r < 8; ++r) {
      int m = mbase + r;
      if (m < N) {
        float v = (acc[t][r] + bb) * sc + sh;
        outH[(size_t)m * HID + n] = (_Float16)fmaxf(v, 0.0f);
      }
    }
  }
}

// -------------------------------------------- WMMA linear + ReLU (pred MLP)
template <int NOUT>
__global__ __launch_bounds__(256) void linear_relu_wmma_kernel(
    const _Float16* __restrict__ xin, const _Float16* __restrict__ WH,
    const float* __restrict__ bias, _Float16* __restrict__ outH, int N) {
  __shared__ _Float16 sW[NOUT * 32];
  const int lane = threadIdx.x & 31;
  const int wave = threadIdx.x >> 5;
  const int rowbase = blockIdx.x * 128 + wave * 16;
  const int mrow = rowbase + (lane & 15);
  const int mload = (mrow < N) ? mrow : (N - 1);
  const _Float16* xrow = xin + (size_t)mload * HID;
  const int khi = (lane >= 16) ? 8 : 0;
  const int bhi = (lane >= 16) ? 16 : 0;
  const int nlo = lane & 15;
  constexpr int NT = NOUT / 16;

  v8f acc[NT];
#pragma unroll
  for (int t = 0; t < NT; ++t) {
    v8f z = {0.f, 0.f, 0.f, 0.f, 0.f, 0.f, 0.f, 0.f};
    acc[t] = z;
  }

  for (int s = 0; s < 4; ++s) {
    __syncthreads();
    for (int i = threadIdx.x; i < NOUT * 4; i += 256) {
      int n = i >> 2;
      int kk = (i & 3) * 8;
      ((int4*)sW)[i] = *(const int4*)(WH + (size_t)n * HID + s * 32 + kk);
    }
    __syncthreads();
    v16h a;
#pragma unroll
    for (int r = 0; r < 8; ++r) {
      int k = s * 32 + 2 * khi + r * 2;
      v2h hx = *(const v2h*)(xrow + k);
      a[2 * r] = hx[0];
      a[2 * r + 1] = hx[1];
    }
    v16h bf[NT];
#pragma unroll
    for (int t = 0; t < NT; ++t)
      load_bfrag(bf[t], sW + (t * 16 + nlo) * 32 + bhi);
#pragma unroll
    for (int t = 0; t < NT; ++t)
      acc[t] = __builtin_amdgcn_wmma_f32_16x16x32_f16(false, a, false, bf[t],
                                                      (short)0, acc[t], false, false);
  }

  const int mbase = rowbase + ((lane >= 16) ? 8 : 0);
#pragma unroll
  for (int t = 0; t < NT; ++t) {
    int n = t * 16 + nlo;
    float bb = bias[n];
#pragma unroll
    for (int r = 0; r < 8; ++r) {
      int m = mbase + r;
      if (m < N)
        outH[(size_t)m * NOUT + n] = (_Float16)fmaxf(acc[t][r] + bb, 0.0f);
    }
  }
}

// --------------------------------------------------- team node update (tiny)
__global__ void team_update_kernel(
    const float* __restrict__ agg1, const float* __restrict__ deg1,
    const float* __restrict__ Wl1, const float* __restrict__ bl1, const float* __restrict__ Wr1,
    const float* __restrict__ agg2, const float* __restrict__ deg2,
    const float* __restrict__ Wl2, const float* __restrict__ bl2, const float* __restrict__ Wr2,
    const _Float16* __restrict__ xtH,
    const float* __restrict__ g, const float* __restrict__ be,
    const float* __restrict__ mn, const float* __restrict__ vr,
    _Float16* __restrict__ outH) {
  int b = blockIdx.x;   // team node
  int o = threadIdx.x;  // out channel
  float i1 = 1.0f / fmaxf(deg1[b], 1.0f);
  float i2 = 1.0f / fmaxf(deg2[b], 1.0f);
  const float* a1 = agg1 + b * HID;
  const float* a2 = agg2 + b * HID;
  const _Float16* xr = xtH + b * HID;
  float v1 = bl1[o], v2 = bl2[o];
  for (int k = 0; k < HID; ++k) {
    float xt = (float)xr[k];
    v1 += (a1[k] * i1) * Wl1[o * HID + k] + xt * Wr1[o * HID + k];
    v2 += (a2[k] * i2) * Wl2[o * HID + k] + xt * Wr2[o * HID + k];
  }
  float v = 0.5f * (v1 + v2);
  float sc = g[o] * rsqrtf(vr[o] + EPSV);
  v = v * sc + (be[o] - mn[o] * sc);
  outH[b * HID + o] = (_Float16)fmaxf(v, 0.0f);
}

// --------------------------------------------------------- final 64 -> 1 dot
__global__ void final_dot_kernel(const _Float16* __restrict__ h2,
                                 const float* __restrict__ w, const float* __restrict__ b,
                                 float* __restrict__ out, int N) {
  int n = blockIdx.x * 256 + threadIdx.x;
  if (n >= N) return;
  const _Float16* hr = h2 + (size_t)n * 64;
  float v = b[0];
  for (int k = 0; k < 64; ++k) v += (float)hr[k] * w[k];
  out[n] = v;
}

// ======================================================================
extern "C" void kernel_launch(void* const* d_in, const int* in_sizes, int n_in,
                              void* d_out, int out_size, void* d_ws, size_t ws_size,
                              hipStream_t stream) {
  (void)in_sizes; (void)n_in; (void)out_size; (void)ws_size;
  const int NBAT = 500000, NPIT = 100000;
  const int EF = 2000000, ET = 2000000, L = 3;

  const float* x_batter  = (const float*)d_in[0];
  const float* x_pitcher = (const float*)d_in[1];
  const float* x_team    = (const float*)d_in[2];
  const int* faces_src = (const int*)d_in[3];
  const int* faces_dst = (const int*)d_in[4];
  const int* tm_src    = (const int*)d_in[5];
  const int* tm_dst    = (const int*)d_in[6];
  const int* bpf_src   = (const int*)d_in[7];
  const int* bpf_dst   = (const int*)d_in[8];
  const int* ppf_src   = (const int*)d_in[9];
  const int* ppf_dst   = (const int*)d_in[10];

  // stem params: for each of batter/pitcher/team: W, b, bn{gamma,beta,mean,var}
  const float *stW[3], *stb[3], *stbn[3][4];
  int ix = 11;
  for (int t = 0; t < 3; ++t) {
    stW[t] = (const float*)d_in[ix++];
    stb[t] = (const float*)d_in[ix++];
    for (int j = 0; j < 4; ++j) stbn[t][j] = (const float*)d_in[ix++];
  }
  // layers: convs tm,faces,bpf,ppf (Wl,bl,Wr each), bns batter,pitcher,team (g,b,m,v)
  const float* cv[3][4][3];
  const float* bn[3][3][4];
  for (int l = 0; l < L; ++l) {
    for (int c = 0; c < 4; ++c)
      for (int j = 0; j < 3; ++j) cv[l][c][j] = (const float*)d_in[ix++];
    for (int t = 0; t < 3; ++t)
      for (int j = 0; j < 4; ++j) bn[l][t][j] = (const float*)d_in[ix++];
  }
  const float* l1W = (const float*)d_in[ix++]; const float* l1b = (const float*)d_in[ix++];
  const float* l2W = (const float*)d_in[ix++]; const float* l2b = (const float*)d_in[ix++];
  const float* l3W = (const float*)d_in[ix++]; const float* l3b = (const float*)d_in[ix++];

  // ---- workspace carve
  char* base = (char*)d_ws;
  size_t off = 0;
  auto carve = [&](size_t bytes) -> void* {
    void* p = base + off;
    off += (bytes + 255) & ~(size_t)255;
    return p;
  };
  _Float16* xbA = (_Float16*)carve((size_t)NBAT * HID * 2);
  _Float16* xbB = (_Float16*)carve((size_t)NBAT * HID * 2);
  _Float16* xpA = (_Float16*)carve((size_t)NPIT * HID * 2);
  _Float16* xpB = (_Float16*)carve((size_t)NPIT * HID * 2);
  _Float16* xtA = (_Float16*)carve((size_t)NTEAM * HID * 2);
  _Float16* xtB = (_Float16*)carve((size_t)NTEAM * HID * 2);
  float* aggb  = (float*)carve((size_t)NBAT * HID * 4);
  float* aggp  = (float*)carve((size_t)NPIT * HID * 4);
  float* aggt1 = (float*)carve((size_t)NTEAM * HID * 4);
  float* aggt2 = (float*)carve((size_t)NTEAM * HID * 4);
  float* degb  = (float*)carve((size_t)NBAT * 4);
  float* degp  = (float*)carve((size_t)NPIT * 4);
  float* degt1 = (float*)carve((size_t)NTEAM * 4);
  float* degt2 = (float*)carve((size_t)NTEAM * 4);
  _Float16* wh[3][4];  // per layer: tm.Wl, tm.Wr, faces.Wl, faces.Wr (f16)
  for (int l = 0; l < L; ++l)
    for (int j = 0; j < 4; ++j) wh[l][j] = (_Float16*)carve((size_t)HID * HID * 2);
  _Float16* l1WH = (_Float16*)carve((size_t)HID * HID * 2);
  _Float16* l2WH = (_Float16*)carve((size_t)64 * HID * 2);

  const int WB = (HID * HID + 255) / 256;
  // ---- weight conversion to f16
  for (int l = 0; l < L; ++l) {
    f32_to_f16_kernel<<<WB, 256, 0, stream>>>(cv[l][0][0], wh[l][0], HID * HID); // tm.Wl
    f32_to_f16_kernel<<<WB, 256, 0, stream>>>(cv[l][0][2], wh[l][1], HID * HID); // tm.Wr
    f32_to_f16_kernel<<<WB, 256, 0, stream>>>(cv[l][1][0], wh[l][2], HID * HID); // faces.Wl
    f32_to_f16_kernel<<<WB, 256, 0, stream>>>(cv[l][1][2], wh[l][3], HID * HID); // faces.Wr
  }
  f32_to_f16_kernel<<<WB, 256, 0, stream>>>(l1W, l1WH, HID * HID);
  f32_to_f16_kernel<<<(64 * HID + 255) / 256, 256, 0, stream>>>(l2W, l2WH, 64 * HID);

  // ---- degrees (edge-structure invariant: once per call)
  hipMemsetAsync(degb, 0, (size_t)NBAT * 4, stream);
  hipMemsetAsync(degp, 0, (size_t)NPIT * 4, stream);
  count_kernel<<<(ET + 255) / 256, 256, 0, stream>>>(tm_dst, degb, ET);
  count_kernel<<<(EF + 255) / 256, 256, 0, stream>>>(faces_dst, degp, EF);

  // ---- stems
  stem_kernel<<<((size_t)NBAT * HID + 255) / 256, 256, 0, stream>>>(
      x_batter, 4, stW[0], stb[0], stbn[0][0], stbn[0][1], stbn[0][2], stbn[0][3], xbA, NBAT);
  stem_kernel<<<((size_t)NPIT * HID + 255) / 256, 256, 0, stream>>>(
      x_pitcher, 2, stW[1], stb[1], stbn[1][0], stbn[1][1], stbn[1][2], stbn[1][3], xpA, NPIT);
  stem_kernel<<<((size_t)NTEAM * HID + 255) / 256, 256, 0, stream>>>(
      x_team, 2, stW[2], stb[2], stbn[2][0], stbn[2][1], stbn[2][2], stbn[2][3], xtA, NTEAM);

  _Float16 *xb = xbA, *xbn = xbB, *xp = xpA, *xpn = xpB, *xt = xtA, *xtn = xtB;

  for (int l = 0; l < L; ++l) {
    hipMemsetAsync(aggb, 0, (size_t)NBAT * HID * 4, stream);
    hipMemsetAsync(aggp, 0, (size_t)NPIT * HID * 4, stream);
    hipMemsetAsync(aggt1, 0, (size_t)NTEAM * HID * 4, stream);
    hipMemsetAsync(aggt2, 0, (size_t)NTEAM * HID * 4, stream);
    hipMemsetAsync(degt1, 0, (size_t)NTEAM * 4, stream);
    hipMemsetAsync(degt2, 0, (size_t)NTEAM * 4, stream);

    scatter_kernel<<<(ET + 7) / 8, 256, 0, stream>>>(xb, tm_src, tm_dst, aggb, ET);
    scatter_kernel<<<(EF + 7) / 8, 256, 0, stream>>>(xb, faces_src, faces_dst, aggp, EF);
    scatter_team_kernel<<<240, 256, 0, stream>>>(xb, bpf_src, bpf_dst, aggt1, degt1, NBAT);
    scatter_team_kernel<<<240, 256, 0, stream>>>(xp, ppf_src, ppf_dst, aggt2, degt2, NPIT);

    sage_wmma_kernel<<<(NBAT + 127) / 128, 256, 0, stream>>>(
        aggb, degb, xb, wh[l][0], wh[l][1], cv[l][0][1],
        bn[l][0][0], bn[l][0][1], bn[l][0][2], bn[l][0][3], xbn, NBAT);
    sage_wmma_kernel<<<(NPIT + 127) / 128, 256, 0, stream>>>(
        aggp, degp, xp, wh[l][2], wh[l][3], cv[l][1][1],
        bn[l][1][0], bn[l][1][1], bn[l][1][2], bn[l][1][3], xpn, NPIT);
    team_update_kernel<<<NTEAM, HID, 0, stream>>>(
        aggt1, degt1, cv[l][2][0], cv[l][2][1], cv[l][2][2],
        aggt2, degt2, cv[l][3][0], cv[l][3][1], cv[l][3][2],
        xt, bn[l][2][0], bn[l][2][1], bn[l][2][2], bn[l][2][3], xtn);

    _Float16* tswp;
    tswp = xb; xb = xbn; xbn = tswp;
    tswp = xp; xp = xpn; xpn = tswp;
    tswp = xt; xt = xtn; xtn = tswp;
  }

  // ---- prediction MLP (agg buffers are free now; reuse aggb as f16 scratch)
  _Float16* h1 = (_Float16*)aggb;
  _Float16* h2 = h1 + (size_t)NBAT * HID;
  linear_relu_wmma_kernel<128><<<(NBAT + 127) / 128, 256, 0, stream>>>(xb, l1WH, l1b, h1, NBAT);
  linear_relu_wmma_kernel<64><<<(NBAT + 127) / 128, 256, 0, stream>>>(h1, l2WH, l2b, h2, NBAT);
  final_dot_kernel<<<(NBAT + 255) / 256, 256, 0, stream>>>(h2, l3W, l3b, (float*)d_out, NBAT);
}
